// LayerMaskedAttention_52742198395223
// MI455X (gfx1250) — matrix operand order repaired and checked by reading, not compiled, not run on hardware
//
#include <hip/hip_runtime.h>

typedef __attribute__((ext_vector_type(16))) _Float16 v16h;
typedef __attribute__((ext_vector_type(8)))  float    v8f;

namespace {
constexpr int kB = 2, kH = 12, kT = 4096, kD = 64;
constexpr int BM    = 128; // query rows per block (8 waves x 16)
constexpr int NWAVE = BM / 16;
constexpr int NTHR  = NWAVE * 32;
constexpr int KTile = 32;  // keys per tile
constexpr int KPAD  = 68;  // fp32 K/V tile row stride (floats); 16B-aligned rows
constexpr int PROW  = 34;  // f16 P staging row stride (halves)
constexpr float kLog2e = 1.4426950408889634f;
}

// One lane copies 16 bytes global -> LDS via the CDNA5 async path (ASYNCcnt).
__device__ __forceinline__ void async_copy16(unsigned lds_off, const void* gaddr) {
  asm volatile("global_load_async_to_lds_b128 %0, %1, off"
               :: "v"(lds_off), "v"((unsigned long long)(uintptr_t)gaddr)
               : "memory");
}

__global__ __launch_bounds__(NTHR)
void fattn_local_kernel(const float* __restrict__ Q,
                        const float* __restrict__ K,
                        const float* __restrict__ V,
                        const int* __restrict__ winp,
                        float* __restrict__ Out) {
  __shared__ float    Kf[KTile * KPAD];        // [key][d] fp32 (async-staged)
  __shared__ float    Vf[KTile * KPAD];        // [key][d] fp32 (async-staged)
  __shared__ _Float16 Ps[NWAVE * 16 * PROW];   // per-wave P transpose scratch

  const int W    = winp[0];
  const int tid  = threadIdx.x;
  const int lane = tid & 31;
  const int wave = tid >> 5;
  const int n16  = lane & 15;   // N / column index within 16
  const int hh   = lane >> 4;   // half-wave select

  const int qtiles = kT / BM;
  const int bh     = blockIdx.x / qtiles;
  const int qs     = (blockIdx.x % qtiles) * BM;
  const int head   = bh % kH;
  const int batch  = bh / kH;

  const float* Qb = Q + (size_t)bh * kT * kD;
  const float* Kb = K + (size_t)bh * kT * kD;
  const float* Vb = V + (size_t)bh * kT * kD;

  const int wq = qs + wave * 16;  // this wave's first query row

  // ---- Q A-fragments (16x32 halves each, K=0..31 / 32..63), pre-scaled ----
  // A layout: lane L: M = L%16, K = (L/16)*16 + 2r + e
  v16h aq0, aq1;
  {
    const float* qrow = Qb + (size_t)(wq + n16) * kD;
    #pragma unroll
    for (int r = 0; r < 8; ++r) {
      const int d = (hh << 4) + (r << 1);
      float2 x0 = *(const float2*)(qrow + d);
      float2 x1 = *(const float2*)(qrow + d + 32);
      auto p0 = __builtin_amdgcn_cvt_pkrtz(x0.x * 0.125f, x0.y * 0.125f);
      auto p1 = __builtin_amdgcn_cvt_pkrtz(x1.x * 0.125f, x1.y * 0.125f);
      aq0[2*r] = p0.x; aq0[2*r+1] = p0.y;
      aq1[2*r] = p1.x; aq1[2*r+1] = p1.y;
    }
  }

  v8f o[4] = {};                 // O accum: 4 chunks of 16 d-cols, rows r+8*hh
  float mrow[8], lrow[8];
  #pragma unroll
  for (int r = 0; r < 8; ++r) { mrow[r] = -1e30f; lrow[r] = 0.0f; }

  int kt0 = qs - (W - 1);
  if (kt0 < 0) kt0 = 0;
  kt0 &= ~(KTile - 1);
  const int ktEnd = qs + BM - 1;  // last (inclusive) key any row in block sees

  for (int kt = kt0; kt <= ktEnd; kt += KTile) {
    __syncthreads();  // protect LDS tiles from previous iteration's readers
    // ---- async stage: raw fp32 K/V tiles, 16B per lane per instruction ----
    for (int c = tid; c < (KTile * kD) / 4; c += NTHR) {
      const int row = c >> 4;            // key within tile
      const int col = (c & 15) << 2;     // d (float4 granule)
      const size_t g = (size_t)(kt + row) * kD + col;
      const unsigned loff = (unsigned)((row * KPAD + col) * 4);
      async_copy16((unsigned)(uintptr_t)(&Kf[0]) + loff, Kb + g);
      async_copy16((unsigned)(uintptr_t)(&Vf[0]) + loff, Vb + g);
    }
    // speculative prefetch of the next key tile (global_prefetch_b8)
    if (tid < KTile && kt + KTile <= ktEnd) {
      int prow = kt + KTile + tid; if (prow >= kT) prow = kT - 1;
      __builtin_prefetch(Kb + (size_t)prow * kD, 0, 1);
      __builtin_prefetch(Vb + (size_t)prow * kD, 0, 1);
    }
    asm volatile("s_wait_asynccnt 0" ::: "memory");  // this wave's copies landed
    __syncthreads();                                  // everyone's copies landed

    // wave-level band overlap test (uniform per wave -> no divergence in wave)
    const bool active = (kt <= wq + 15) && (kt + KTile - 1 >= wq - W + 1);
    if (!active) continue;

    // ---- S = Q * K^T over 32 keys: two 16-wide key subtiles, K-dim 64 ----
    // B layout: lane L: N(key) = L%16, K(d) = (L/16)*16 + 2r + e
    v8f s[2];
    #pragma unroll
    for (int sub = 0; sub < 2; ++sub) {
      v16h b0, b1;
      const float* kp = &Kf[(sub * 16 + n16) * KPAD + (hh << 4)];
      #pragma unroll
      for (int r = 0; r < 8; ++r) {
        float2 x0 = *(const float2*)(kp + 2*r);        // d = hh*16 + 2r (+e)
        float2 x1 = *(const float2*)(kp + 2*r + 32);   // second K-dim fragment
        auto h0 = __builtin_amdgcn_cvt_pkrtz(x0.x, x0.y);
        auto h1 = __builtin_amdgcn_cvt_pkrtz(x1.x, x1.y);
        b0[2*r] = h0.x; b0[2*r+1] = h0.y;
        b1[2*r] = h1.x; b1[2*r+1] = h1.y;
      }
      v8f c = {};
      c = __builtin_amdgcn_wmma_f32_16x16x32_f16(false, aq0, false, b0, (short)0, c, false, false);
      c = __builtin_amdgcn_wmma_f32_16x16x32_f16(false, aq1, false, b1, (short)0, c, false, false);
      s[sub] = c;
    }

    // ---- online softmax: C/D layout row M = r + 8*hh, col N = n16 ----
    float p0[8], p1[8];
    #pragma unroll
    for (int r = 0; r < 8; ++r) {
      const int qi = wq + r + (hh << 3);
      const int k0 = kt + n16;
      const int k1 = kt + 16 + n16;
      const bool v0 = (k0 <= qi) && (k0 > qi - W);
      const bool v1 = (k1 <= qi) && (k1 > qi - W);
      float s0 = v0 ? s[0][r] : -1e30f;
      float s1 = v1 ? s[1][r] : -1e30f;
      float mx = fmaxf(s0, s1);
      mx = fmaxf(mx, __shfl_xor(mx, 1));
      mx = fmaxf(mx, __shfl_xor(mx, 2));
      mx = fmaxf(mx, __shfl_xor(mx, 4));
      mx = fmaxf(mx, __shfl_xor(mx, 8));       // row reduce within 16-lane half
      const float mnew  = fmaxf(mrow[r], mx);
      const float alpha = exp2f((mrow[r] - mnew) * kLog2e);
      const float e0 = v0 ? exp2f((s0 - mnew) * kLog2e) : 0.0f;  // masked -> 0
      const float e1 = v1 ? exp2f((s1 - mnew) * kLog2e) : 0.0f;
      float rs = e0 + e1;
      rs += __shfl_xor(rs, 1);
      rs += __shfl_xor(rs, 2);
      rs += __shfl_xor(rs, 4);
      rs += __shfl_xor(rs, 8);
      lrow[r] = lrow[r] * alpha + rs;
      mrow[r] = mnew;
      #pragma unroll
      for (int c4 = 0; c4 < 4; ++c4) o[c4][r] *= alpha;
      p0[r] = e0; p1[r] = e1;
    }

    // ---- transpose P (C-layout) -> A-layout via per-wave LDS scratch ----
    _Float16* pw = &Ps[wave * 16 * PROW];
    #pragma unroll
    for (int r = 0; r < 8; ++r) {
      const int row = r + (hh << 3);
      pw[row * PROW + n16]      = (_Float16)p0[r];
      pw[row * PROW + 16 + n16] = (_Float16)p1[r];
    }
    // same-wave DS ops are in-order; wait for the stores + block reordering
    asm volatile("s_wait_dscnt 0" ::: "memory");
    v16h pa;
    {
      const _Float16* pr = &pw[n16 * PROW + (hh << 3)];
      #pragma unroll
      for (int r = 0; r < 8; ++r) {
        const int kk = ((r >> 2) << 4) + ((r & 3) << 1);
        pa[2*r]   = pr[kk];
        pa[2*r+1] = pr[kk + 1];
      }
    }

    // ---- O += P * V : B[k=key][n=d-col], key rows straight from fp32 tile ----
    #pragma unroll
    for (int c4 = 0; c4 < 4; ++c4) {
      v16h vb;
      #pragma unroll
      for (int r = 0; r < 8; ++r) {
        const float* vp = &Vf[(((r >> 2) << 4) + ((r & 3) << 1) + (hh << 3)) * KPAD + c4 * 16 + n16];
        auto hv = __builtin_amdgcn_cvt_pkrtz(vp[0], vp[KPAD]);  // two adjacent keys (vp row and the next row)
        vb[2*r] = hv.x; vb[2*r+1] = hv.y;
      }
      o[c4] = __builtin_amdgcn_wmma_f32_16x16x32_f16(false, pa, false, vb, (short)0, o[c4], false, false);
    }
  }

  // ---- normalize + store: out[batch][t][head*64 + d] ----
  float* ob = Out + ((size_t)batch * kT) * (kH * kD) + head * kD;
  #pragma unroll
  for (int r = 0; r < 8; ++r) {
    const int row = r + (hh << 3);
    const float inv = 1.0f / lrow[r];   // every row has >= 1 valid key (j == i)
    float* orow = ob + (size_t)(wq + row) * (kH * kD);
    #pragma unroll
    for (int c4 = 0; c4 < 4; ++c4)
      orow[c4 * 16 + n16] = o[c4][r] * inv;
  }
}

extern "C" void kernel_launch(void* const* d_in, const int* in_sizes, int n_in,
                              void* d_out, int out_size, void* d_ws, size_t ws_size,
                              hipStream_t stream) {
  (void)in_sizes; (void)n_in; (void)out_size; (void)d_ws; (void)ws_size;
  const float* q = (const float*)d_in[0];
  const float* k = (const float*)d_in[1];
  const float* v = (const float*)d_in[2];
  const int*   w = (const int*)d_in[3];
  float* out = (float*)d_out;
  dim3 grid(kB * kH * (kT / BM));
  fattn_local_kernel<<<grid, dim3(NTHR), 0, stream>>>(q, k, v, w, out);
}
